// TagSpecCtxGenerator_69801808495268
// MI455X (gfx1250) — matrix operand order repaired and checked by reading, not compiled, not run on hardware
//
#include <hip/hip_runtime.h>
#include <hip/hip_bf16.h>

// ---------------------------------------------------------------------------
// Problem constants (from reference)
// ---------------------------------------------------------------------------
#define BB 32
#define LL 256
#define TT (BB * LL)        // 8192 tokens
#define DD 256              // WED + PED
#define WED 200
#define PED 56
#define HH 512
#define H4 2048
#define MM 1024             // per-tag vocab
#define EE 100
#define NP 128              // padded E for GEMM
#define NTAGS 16

typedef __bf16 v16bf __attribute__((ext_vector_type(16)));
typedef float  v8f   __attribute__((ext_vector_type(8)));

__device__ __forceinline__ v8f wmma_bf16(v16bf a, v16bf b, v8f c) {
    return __builtin_amdgcn_wmma_f32_16x16x32_bf16(
        /*neg_a=*/false, a, /*neg_b=*/false, b,
        /*c_mod=*/(short)0, c, /*reuse_a=*/false, /*reuse_b=*/false);
}

// A fragment: 16x32 bf16 tile, row-major with stride lda.
// ISA layout: lanes 0-15 -> M=lane, K base 0; lanes 16-31 -> M=lane-16, K base 8.
// VGPR i holds K = kbase + (i>=4?16:0) + (i&3)*2, pairs contiguous.
__device__ __forceinline__ v16bf load_a_frag(const __bf16* Ap, int lda, int lane) {
    int m  = lane & 15;
    int kb = (lane & 16) ? 8 : 0;
    const __bf16* p = Ap + (size_t)m * lda + kb;
    v16bf r;
#pragma unroll
    for (int i = 0; i < 8; ++i) {
        int k = ((i & 4) << 2) + ((i & 3) << 1);
        r[2 * i]     = p[k];
        r[2 * i + 1] = p[k + 1];
    }
    return r;
}

// B fragment: 32x16 bf16 tile supplied as BT[n][k] (N-major), stride ldb.
// lanes 0-15 -> K base 0, lanes 16-31 -> K base 8; VGPR j -> K = kbase + (j>=4?16:0) + (j&3)*2, pairs contiguous.
__device__ __forceinline__ v16bf load_b_frag(const __bf16* BTp, int ldb, int lane) {
    int n = lane & 15;
    const __bf16* p = BTp + (size_t)n * ldb + ((lane & 16) ? 8 : 0);
    v16bf r;
#pragma unroll
    for (int j = 0; j < 8; ++j) {
        int k = ((j & 4) << 2) + ((j & 3) << 1);
        r[2 * j]     = p[k];
        r[2 * j + 1] = p[k + 1];
    }
    return r;
}

__device__ __forceinline__ float sigmoidf(float x) { return 1.0f / (1.0f + expf(-x)); }

// ---------------------------------------------------------------------------
// Prep kernels: convert + transpose weights to bf16 [N][K]
// ---------------------------------------------------------------------------
__global__ void transpose_cvt(const float* __restrict__ src, __bf16* __restrict__ dst,
                              int K, int N) {
    size_t base = (size_t)blockIdx.z * K * N;
    int idx = blockIdx.x * blockDim.x + threadIdx.x;
    if (idx >= K * N) return;
    int k = idx / N, n = idx - k * N;
    dst[base + (size_t)n * K + k] = (__bf16)src[base + idx];
}

// Pt[p][n][k] = (n < E) ? psr[words_tbl[p][k]][n] : 0   (padded to NP=128)
__global__ void build_P(const float* __restrict__ psr, const int* __restrict__ wt,
                        __bf16* __restrict__ Pt) {
    int idx = blockIdx.x * 256 + threadIdx.x;      // 16*128*1024 total
    int p   = idx >> 17;
    int rem = idx & 131071;
    int n   = rem >> 10;
    int k   = rem & 1023;
    float v = 0.0f;
    if (n < EE) v = psr[(size_t)wt[p * MM + k] * EE + n];
    Pt[idx] = (__bf16)v;
}

// X[t][0:200]=word_emb[w], X[t][200:256]=pos_emb[p], as bf16
__global__ void embed_kernel(const int* __restrict__ w, const int* __restrict__ p,
                             const float* __restrict__ we, const float* __restrict__ pe,
                             __bf16* __restrict__ X) {
    int t = blockIdx.x, d = threadIdx.x;
    float v = (d < WED) ? we[(size_t)w[t] * WED + d] : pe[(size_t)p[t] * PED + (d - WED)];
    X[(size_t)t * DD + d] = (__bf16)v;
}

__global__ void init_kernel(int* cnt, float* ent) {
    int i = threadIdx.x;
    if (i < NTAGS) { cnt[i] = 0; ent[i] = 0.0f; }
}

// counting sort tokens by tag
__global__ void route_kernel(const int* __restrict__ pos, int* cnt, int* rows) {
    int t = blockIdx.x * 256 + threadIdx.x;
    if (t >= TT) return;
    int p = pos[t];
    int s = atomicAdd(&cnt[p], 1);
    rows[p * TT + s] = t;
}

// ---------------------------------------------------------------------------
// Generic gathered bf16 WMMA GEMM:  C[row][*] = A[row] @ BT^T + bias
// A: [*, K] bf16 row-major (rows gathered via rmap), BT: [N][K] bf16,
// C: f32 with leading dim ldc, scattered to the same gathered row index.
// Block: 128 threads (4 waves), tile 64(M) x 128(N), K step 32.
// Tiles are staged into LDS with CDNA5 GLOBAL_LOAD_ASYNC_TO_LDS_B128
// (ASYNCcnt-tracked); out-of-range gathered rows are zero-filled with
// ds_store to disjoint addresses.
// ---------------------------------------------------------------------------
#define GT_BK 32
#define GT_ABYTES (64 * GT_BK * 2)      // 4 KB A tile
#define GT_BBYTES (128 * GT_BK * 2)     // 8 KB B tile
#define GT_LDS (GT_ABYTES + GT_BBYTES)

__global__ __launch_bounds__(128)
void gemm_bf16_kernel(const __bf16* __restrict__ A, const __bf16* __restrict__ BT,
                      float* __restrict__ C, const float* __restrict__ bias,
                      const int* __restrict__ rows, const int* __restrict__ cnts,
                      int Mfull, int Nstore, int K, int ldc,
                      long long strideBT, long long strideBias, int strideRows) {
    int z = blockIdx.z;
    BT += (size_t)z * strideBT;
    if (bias) bias += (size_t)z * strideBias;
    const int* rmap = rows ? rows + (size_t)z * strideRows : nullptr;
    int Mz = cnts ? cnts[z] : Mfull;
    int row0 = blockIdx.x * 64;
    if (row0 >= Mz) return;
    int n0 = blockIdx.y * 128;

    extern __shared__ __align__(16) char smemraw[];
    __bf16* At = (__bf16*)smemraw;                    // [64][32]
    __bf16* Bt = (__bf16*)(smemraw + GT_ABYTES);      // [128][32]
    // LDS byte offset of the dynamic-shared base (for the async-copy asm)
    unsigned ldsbase = __builtin_amdgcn_groupstaticsize();

    int tid = threadIdx.x;
    int wave = tid >> 5, lane = tid & 31;

    v8f acc[8] = {};

    for (int k0 = 0; k0 < K; k0 += GT_BK) {
        // ---- stage A tile (64x32): async global->LDS, zero-fill OOB rows ----
#pragma unroll
        for (int c = 0; c < 2; ++c) {
            int chunk = tid + c * 128;              // 0..255 : 64 rows x 4 x 16B
            int r = chunk >> 2, seg = chunk & 3;
            int ar = row0 + r;
            if (ar < Mz) {
                int grow = rmap ? rmap[ar] : ar;
                const void* gp = (const void*)(A + (size_t)grow * K + k0 + seg * 8);
                unsigned ldsoff = ldsbase + (unsigned)(chunk * 16);
                asm volatile("global_load_async_to_lds_b128 %0, %1, off"
                             :: "v"(ldsoff), "v"(gp) : "memory");
            } else {
                uint4 zz = {0u, 0u, 0u, 0u};
                ((uint4*)At)[chunk] = zz;
            }
        }
        // ---- stage B tile (128x32): async global->LDS ----
#pragma unroll
        for (int c = 0; c < 4; ++c) {
            int chunk = tid + c * 128;              // 0..511 : 128 rows x 4 x 16B
            int r = chunk >> 2, seg = chunk & 3;
            const void* gp = (const void*)(BT + (size_t)(n0 + r) * K + k0 + seg * 8);
            unsigned ldsoff = ldsbase + (unsigned)(GT_ABYTES + chunk * 16);
            asm volatile("global_load_async_to_lds_b128 %0, %1, off"
                         :: "v"(ldsoff), "v"(gp) : "memory");
        }
        if (k0 + GT_BK < K)
            __builtin_prefetch((const void*)(BT + (size_t)(n0 + tid) * K + k0 + GT_BK), 0, 1);
        // drain this wave's async copies, then rendezvous with the other waves
        asm volatile("s_wait_asynccnt 0x0" ::: "memory");
        __syncthreads();

        v16bf a = load_a_frag(At + wave * 16 * GT_BK, GT_BK, lane);
#pragma unroll
        for (int s = 0; s < 8; ++s) {
            v16bf b = load_b_frag(Bt + s * 16 * GT_BK, GT_BK, lane);
            acc[s] = wmma_bf16(a, b, acc[s]);
        }
        __syncthreads();
    }

    int n_lane = lane & 15;
    int mhi = (lane & 16) ? 8 : 0;
#pragma unroll
    for (int s = 0; s < 8; ++s) {
        int col = n0 + s * 16 + n_lane;
        if (col >= Nstore) continue;
        float bv = bias ? bias[col] : 0.0f;
#pragma unroll
        for (int v = 0; v < 8; ++v) {
            int ridx = row0 + wave * 16 + v + mhi;
            if (ridx < Mz) {
                int grow = rmap ? rmap[ridx] : ridx;
                C[(size_t)grow * ldc + col] = acc[s][v] + bv;
            }
        }
    }
}

// ---------------------------------------------------------------------------
// Bidirectional LSTM scan. grid.x=2 (dir), 1024 threads (32 waves).
// h kept in LDS (bf16, WMMA A operand), c kept in registers in the exact
// C-fragment lane mapping. Wave w owns hidden units [16w, 16w+16) and computes
// all four gate column tiles for them, so i/f/g/o of a unit live in one lane.
// ---------------------------------------------------------------------------
__global__ __launch_bounds__(1024)
void lstm_scan(const __bf16* __restrict__ WhT_f, const __bf16* __restrict__ WhT_b,
               const float* __restrict__ Gf, const float* __restrict__ Gb,
               const float* __restrict__ mask, __bf16* __restrict__ ctx) {
    __shared__ __bf16 hS[BB * HH];   // 32 KB: h[batch][unit]

    int dir = blockIdx.x;
    const __bf16* WhT = dir ? WhT_b : WhT_f;
    const float*  G   = dir ? Gb : Gf;
    int coloff = dir * HH;

    int tid = threadIdx.x;
    int wave = tid >> 5, lane = tid & 31;
    int u0 = wave * 16;

    for (int i = tid; i < BB * HH; i += 1024) hS[i] = (__bf16)0.0f;

    // per-lane cell state for the 16 (batch,unit) cells this lane owns
    float creg[16];
#pragma unroll
    for (int i = 0; i < 16; ++i) creg[i] = 0.0f;

    int n_lane = lane & 15;
    int mhi = (lane & 16) ? 8 : 0;
    int u = u0 + n_lane;

    __syncthreads();

    for (int s = 0; s < LL; ++s) {
        int t = dir ? (LL - 1 - s) : s;

        v8f acc[4][2] = {};
        for (int k0 = 0; k0 < HH; k0 += 32) {
            v16bf a0 = load_a_frag(hS + 0 * HH + k0, HH, lane);   // batches 0-15
            v16bf a1 = load_a_frag(hS + 16 * HH + k0, HH, lane);  // batches 16-31
#pragma unroll
            for (int g = 0; g < 4; ++g) {
                const __bf16* bp = WhT + (size_t)(g * HH + u0) * HH + k0;
                v16bf b = load_b_frag(bp, HH, lane);
                acc[g][0] = wmma_bf16(a0, b, acc[g][0]);
                acc[g][1] = wmma_bf16(a1, b, acc[g][1]);
            }
        }
        __syncthreads();   // all hS reads done before update

#pragma unroll
        for (int mt = 0; mt < 2; ++mt) {
#pragma unroll
            for (int v = 0; v < 8; ++v) {
                int batch = mt * 16 + v + mhi;
                size_t tok = (size_t)batch * LL + t;
                const float* gx = G + tok * H4;
                float ig = acc[0][mt][v] + gx[0 * HH + u];
                float fg = acc[1][mt][v] + gx[1 * HH + u];
                float gg = acc[2][mt][v] + gx[2 * HH + u];
                float og = acc[3][mt][v] + gx[3 * HH + u];
                float iv = sigmoidf(ig), fv = sigmoidf(fg);
                float gv = tanhf(gg),   ov = sigmoidf(og);
                float cold = creg[mt * 8 + v];
                float hold = (float)hS[batch * HH + u];
                float cn = fv * cold + iv * gv;
                float hn = ov * tanhf(cn);
                float mv = mask[tok];
                hn = hn * mv + hold * (1.0f - mv);
                cn = cn * mv + cold * (1.0f - mv);
                creg[mt * 8 + v] = cn;
                hS[batch * HH + u] = (__bf16)hn;
                ctx[tok * (2 * HH) + coloff + u] = (__bf16)hn;
            }
        }
        __syncthreads();   // updates visible before next step's reads
    }
}

// ---------------------------------------------------------------------------
// Fused softmax / entropy / gumbel-softmax per token row (1024 logits).
// softmax((logp+g)/1) == softmax(logits+g); entropy needs lse.
// ---------------------------------------------------------------------------
__global__ __launch_bounds__(256)
void softmax_gumbel(const float* __restrict__ logits, const float* __restrict__ gum,
                    const int* __restrict__ pos, const int* __restrict__ wtbl,
                    __bf16* __restrict__ spt, int* __restrict__ oword,
                    float* __restrict__ ent_sum) {
    int row = blockIdx.x, tid = threadIdx.x;
    const float* Lr = logits + (size_t)row * MM;
    const float* Gr = gum + (size_t)row * MM;
    __shared__ float sm[256];
    __shared__ int   si[256];

    // max(logits)
    float mx = -1e30f;
    for (int e = tid; e < MM; e += 256) mx = fmaxf(mx, Lr[e]);
    sm[tid] = mx; __syncthreads();
    for (int s = 128; s > 0; s >>= 1) { if (tid < s) sm[tid] = fmaxf(sm[tid], sm[tid + s]); __syncthreads(); }
    mx = sm[0]; __syncthreads();

    // lse
    float se = 0.0f;
    for (int e = tid; e < MM; e += 256) se += expf(Lr[e] - mx);
    sm[tid] = se; __syncthreads();
    for (int s = 128; s > 0; s >>= 1) { if (tid < s) sm[tid] += sm[tid + s]; __syncthreads(); }
    float lse = mx + logf(sm[0]); __syncthreads();

    // entropy contribution sum(-logp * p)
    float ent = 0.0f;
    for (int e = tid; e < MM; e += 256) { float lp = Lr[e] - lse; ent -= lp * expf(lp); }
    sm[tid] = ent; __syncthreads();
    for (int s = 128; s > 0; s >>= 1) { if (tid < s) sm[tid] += sm[tid + s]; __syncthreads(); }
    if (tid == 0) atomicAdd(&ent_sum[pos[row]], sm[0]);
    __syncthreads();

    // gumbel perturbed: max + argmax
    float my = -1e30f; int ai = 0;
    for (int e = tid; e < MM; e += 256) {
        float g = -logf(-logf(Gr[e] + 1e-10f) + 1e-10f);
        float y = Lr[e] + g;
        if (y > my) { my = y; ai = e; }
    }
    sm[tid] = my; si[tid] = ai; __syncthreads();
    for (int s = 128; s > 0; s >>= 1) {
        if (tid < s && sm[tid + s] > sm[tid]) { sm[tid] = sm[tid + s]; si[tid] = si[tid + s]; }
        __syncthreads();
    }
    my = sm[0]; int amax = si[0]; __syncthreads();

    float sy = 0.0f;
    for (int e = tid; e < MM; e += 256) {
        float g = -logf(-logf(Gr[e] + 1e-10f) + 1e-10f);
        sy += expf(Lr[e] + g - my);
    }
    sm[tid] = sy; __syncthreads();
    for (int s = 128; s > 0; s >>= 1) { if (tid < s) sm[tid] += sm[tid + s]; __syncthreads(); }
    float inv = 1.0f / sm[0];

    for (int e = tid; e < MM; e += 256) {
        float g = -logf(-logf(Gr[e] + 1e-10f) + 1e-10f);
        spt[(size_t)row * MM + e] = (__bf16)(expf(Lr[e] + g - my) * inv);
    }
    if (tid == 0) oword[row] = wtbl[pos[row] * MM + amax];
}

__global__ void finalize_kernel(const int* __restrict__ ow_cand, const int* __restrict__ w,
                                const int* __restrict__ pos, const float* __restrict__ psr,
                                float* out_word, float* out_emb, float* out_mask,
                                float* out_pri) {
    int t = blockIdx.x * 256 + threadIdx.x;
    if (t >= TT) return;
    int ow = ow_cand[t];
    bool avoid = (ow == w[t]);     // obf_mask is always true (pos < N_TAGS)
    if (avoid) ow = 0;             // UNK_ID
    out_word[t] = (float)ow;
    out_mask[t] = 1.0f;
    out_pri[t]  = (pos[t] < 4) ? 1.0f : 0.0f;
    if (avoid) {
        for (int e = 0; e < EE; ++e) out_emb[(size_t)t * EE + e] = psr[e];
    }
}

__global__ void loss_kernel(const int* cnt, const float* ent_sum, float* out) {
    if (threadIdx.x == 0 && blockIdx.x == 0) {
        float s = 0.0f;
        for (int p = 0; p < NTAGS; ++p) {
            int n = cnt[p];
            if (n > 0) s += ent_sum[p] / fmaxf((float)n * (float)MM, 1.0f);
        }
        *out = -0.01f * s;
    }
}

// ---------------------------------------------------------------------------
extern "C" void kernel_launch(void* const* d_in, const int* in_sizes, int n_in,
                              void* d_out, int out_size, void* d_ws, size_t ws_size,
                              hipStream_t stream) {
    (void)in_sizes; (void)n_in; (void)out_size; (void)ws_size;
    const int*   inp_word = (const int*)d_in[0];
    const int*   inp_pos  = (const int*)d_in[1];
    const float* inp_mask = (const float*)d_in[2];
    const float* gumbel_u = (const float*)d_in[3];
    const float* word_emb = (const float*)d_in[4];
    const float* pos_emb  = (const float*)d_in[5];
    const float* Wx_f     = (const float*)d_in[6];
    const float* Wh_f     = (const float*)d_in[7];
    const float* b_f      = (const float*)d_in[8];
    const float* Wx_b     = (const float*)d_in[9];
    const float* Wh_b     = (const float*)d_in[10];
    const float* b_b      = (const float*)d_in[11];
    const float* dec_W    = (const float*)d_in[12];
    const float* dec_b    = (const float*)d_in[13];
    const int*   words_tbl= (const int*)d_in[14];
    const float* psr      = (const float*)d_in[15];

    float* out       = (float*)d_out;
    float* out_word  = out;
    float* out_emb   = out + TT;
    float* out_mask  = out + TT + TT * EE;
    float* out_pri   = out_mask + TT;
    float* out_loss  = out_pri + TT;

    // workspace bump allocator
    char* ws = (char*)d_ws; size_t off = 0;
    auto alloc = [&](size_t bytes) -> void* {
        void* p = ws + off; off = (off + bytes + 255) & ~(size_t)255; return p;
    };
    __bf16* WxT_f  = (__bf16*)alloc((size_t)H4 * DD * 2);
    __bf16* WxT_b  = (__bf16*)alloc((size_t)H4 * DD * 2);
    __bf16* WhT_f  = (__bf16*)alloc((size_t)H4 * HH * 2);
    __bf16* WhT_b  = (__bf16*)alloc((size_t)H4 * HH * 2);
    __bf16* decWT  = (__bf16*)alloc((size_t)NTAGS * MM * MM * 2);
    __bf16* Pt     = (__bf16*)alloc((size_t)NTAGS * NP * MM * 2);
    __bf16* X      = (__bf16*)alloc((size_t)TT * DD * 2);
    float*  Gf     = (float*)alloc((size_t)TT * H4 * 4);
    float*  Gb     = (float*)alloc((size_t)TT * H4 * 4);
    __bf16* ctx    = (__bf16*)alloc((size_t)TT * 2 * HH * 2);
    float*  logits = (float*)alloc((size_t)TT * MM * 4);
    __bf16* spt    = (__bf16*)alloc((size_t)TT * MM * 2);
    int*    rows   = (int*)alloc((size_t)NTAGS * TT * 4);
    int*    cnt    = (int*)alloc(NTAGS * 4);
    float*  ent    = (float*)alloc(NTAGS * 4);
    int*    owbuf  = (int*)alloc((size_t)TT * 4);

    // --- prep: transpose+convert weights to bf16 [N][K] ---
    transpose_cvt<<<dim3((DD * H4 + 255) / 256, 1, 1), 256, 0, stream>>>(Wx_f, WxT_f, DD, H4);
    transpose_cvt<<<dim3((DD * H4 + 255) / 256, 1, 1), 256, 0, stream>>>(Wx_b, WxT_b, DD, H4);
    transpose_cvt<<<dim3((HH * H4 + 255) / 256, 1, 1), 256, 0, stream>>>(Wh_f, WhT_f, HH, H4);
    transpose_cvt<<<dim3((HH * H4 + 255) / 256, 1, 1), 256, 0, stream>>>(Wh_b, WhT_b, HH, H4);
    transpose_cvt<<<dim3((MM * MM + 255) / 256, 1, NTAGS), 256, 0, stream>>>(dec_W, decWT, MM, MM);
    build_P<<<dim3((NTAGS * NP * MM) / 256, 1, 1), 256, 0, stream>>>(psr, words_tbl, Pt);
    embed_kernel<<<dim3(TT, 1, 1), DD, 0, stream>>>(inp_word, inp_pos, word_emb, pos_emb, X);
    init_kernel<<<1, 32, 0, stream>>>(cnt, ent);
    route_kernel<<<dim3(TT / 256, 1, 1), 256, 0, stream>>>(inp_pos, cnt, rows);

    // --- gates_x = X @ Wx + b, both directions (WMMA, async-LDS staging) ---
    gemm_bf16_kernel<<<dim3(TT / 64, H4 / 128, 1), 128, GT_LDS, stream>>>(
        X, WxT_f, Gf, b_f, nullptr, nullptr, TT, H4, DD, H4, 0, 0, 0);
    gemm_bf16_kernel<<<dim3(TT / 64, H4 / 128, 1), 128, GT_LDS, stream>>>(
        X, WxT_b, Gb, b_b, nullptr, nullptr, TT, H4, DD, H4, 0, 0, 0);

    // --- BiLSTM scan (WMMA recurrence) ---
    lstm_scan<<<dim3(2, 1, 1), 1024, 0, stream>>>(WhT_f, WhT_b, Gf, Gb, inp_mask, ctx);

    // --- per-tag gathered decoder GEMM: logits[t] = ctx[t] @ dec_W[tag] + b ---
    gemm_bf16_kernel<<<dim3(TT / 64, MM / 128, NTAGS), 128, GT_LDS, stream>>>(
        ctx, decWT, logits, dec_b, rows, cnt, TT, MM, MM, MM,
        (long long)MM * MM, (long long)MM, TT);

    // --- fused softmax / entropy / gumbel ---
    softmax_gumbel<<<dim3(TT, 1, 1), 256, 0, stream>>>(
        logits, gumbel_u, inp_pos, words_tbl, spt, owbuf, ent);

    // --- obf_emb = spt @ P[tag] (gathered, N padded to 128, store 100) ---
    gemm_bf16_kernel<<<dim3(TT / 64, NP / 128, NTAGS), 128, GT_LDS, stream>>>(
        spt, Pt, out_emb, nullptr, rows, cnt, TT, EE, MM, EE,
        (long long)NP * MM, 0, TT);

    // --- avoid-collision fixup, masks, entropy loss ---
    finalize_kernel<<<dim3(TT / 256, 1, 1), 256, 0, stream>>>(
        owbuf, inp_word, inp_pos, psr, out_word, out_emb, out_mask, out_pri);
    loss_kernel<<<1, 32, 0, stream>>>(cnt, ent, out_loss);
}
